// SE3TransformerLayer_75522704933008
// MI455X (gfx1250) — matrix operand order repaired and checked by reading, not compiled, not run on hardware
//
#include <hip/hip_runtime.h>
#include <hip/hip_bf16.h>
#include <math.h>

// ---------------------------------------------------------------------------
// SE(3)-Transformer forward for MI455X (gfx1250, wave32).
// Dominant work: per-edge radial-MLP GEMMs -> v_wmma_f32_16x16x32_f16
// (f16 in / f32 accum), fused with the CG message contraction in VALU
// against an LDS-resident factored basis T[o,c,j] = sum_i B[o,i,j] f[c,i].
// w3 weight panels are streamed into LDS with GLOBAL_LOAD_ASYNC_TO_LDS_B32
// (ASYNCcnt) and consumed as WMMA B-fragments from LDS.
// Compute-bound (~250 GFLOP, ~100 MB HBM traffic); weights L2-resident.
// Workspace layout is carved deterministically from d_ws (~52 MB assumed).
// ---------------------------------------------------------------------------

#define N_NODES 2048
#define GRAPHS  4
#define NPG     (N_NODES / GRAPHS)
#define KNN     16
#define NEDGE   (N_NODES * KNN)
#define MID_C   32
#define HEAD_C  8
#define LMAX    6

typedef __attribute__((ext_vector_type(16))) _Float16 v16h;
typedef __attribute__((ext_vector_type(8)))  float    v8f;

// ----------------------------- WMMA helpers --------------------------------
static __device__ __forceinline__ v8f wmma_f16(v16h a, v16h b, v8f c) {
  return __builtin_amdgcn_wmma_f32_16x16x32_f16(false, a, false, b, (short)0, c,
                                                false, false);
}

// A fragment: 16x32 f16, source row-major f32 [16][ld] (LDS).
// Layout (ISA 7.12.2): M = lane&15; K = e<8 ? (lane<16?0:8)+e : (lane<16?16:24)+(e-8)
static __device__ __forceinline__ v16h frag_a_f16(const float* s, int ld) {
  int lane = threadIdx.x & 31;
  int m   = lane & 15;
  int kb0 = (lane < 16) ? 0 : 8;
  int kb1 = (lane < 16) ? 16 : 24;
  v16h a;
#pragma unroll
  for (int e = 0; e < 16; ++e) {
    int k = (e < 8) ? (kb0 + e) : (kb1 + (e - 8));
    a[e] = (_Float16)s[m * ld + k];
  }
  return a;
}

// B fragment: 32x16 f16, source row-major f32 [32][ld] (LDS), tile col0.
// Layout: N = lane&15; K = e<8 ? (lane<16?0:8)+e : (lane<16?16:24)+(e-8)
static __device__ __forceinline__ v16h frag_b_lds(const float* s, int ld, int col0) {
  int lane = threadIdx.x & 31;
  int n  = col0 + (lane & 15);
  int kb = (lane < 16) ? 0 : 8;
  v16h b;
#pragma unroll
  for (int e = 0; e < 16; ++e) b[e] = (_Float16)s[((e < 8) ? (kb + e) : (16 + kb + e - 8)) * ld + n];
  return b;
}

// D store: rows M = r + (lane<16?0:8), col = col0 + (lane&15). Optional bias+relu.
static __device__ __forceinline__ void store_tile(float* s, int ld, v8f c,
                                                  int col0, const float* bias,
                                                  int bias0, int relu, int ncols) {
  int lane = threadIdx.x & 31;
  int n  = lane & 15;
  int mb = (lane < 16) ? 0 : 8;
  int col = col0 + n;
  if (col < ncols) {
    float bv = bias ? bias[bias0 + col] : 0.0f;
#pragma unroll
    for (int r = 0; r < 8; ++r) {
      float v = c[r] + bv;
      if (relu) v = v > 0.0f ? v : 0.0f;
      s[(mb + r) * ld + col] = v;
    }
  }
}

// Async global -> LDS copy (gfx1250 VGLOBAL op, tracked with ASYNCcnt).
// vdst = LDS byte address, vaddr = 32-bit byte offset, saddr = 64-bit base.
static __device__ __forceinline__ void async_g2l_b32(unsigned lds_byte_addr,
                                                     const float* sbase,
                                                     int voff_bytes) {
  asm volatile("global_load_async_to_lds_b32 %0, %1, %2"
               :
               : "v"(lds_byte_addr), "v"(voff_bytes), "s"(sbase)
               : "memory");
}
static __device__ __forceinline__ void wait_asynccnt0() {
  asm volatile("s_wait_asynccnt 0" ::: "memory");
}

static __device__ __forceinline__ int cg_idx(int dout, int di, int J, int o,
                                             int i, int m) {
  return ((((dout * 4 + di) * 7 + J) * 7 + o) * 7 + i) * 13 + m;
}

// ------------------------- CG table construction ---------------------------
struct cpx { double re, im; };
static __device__ __forceinline__ cpx cmul(cpx a, cpx b) {
  cpx r; r.re = a.re * b.re - a.im * b.im; r.im = a.re * b.im + a.im * b.re; return r;
}
static __device__ __forceinline__ cpx cconj(cpx a) { cpx r; r.re = a.re; r.im = -a.im; return r; }

static __device__ double dfact(int n) {
  double f = 1.0;
  for (int i = 2; i <= n; ++i) f *= (double)i;
  return f;
}

static __device__ double su2cg(int j1, int j2, int j3, int m1, int m2, int m3) {
  if (m3 != m1 + m2) return 0.0;
  int vmin = -j1 + j2 + m3;
  if (-j1 + m1 > vmin) vmin = -j1 + m1;
  if (0 > vmin) vmin = 0;
  int vmax = j2 + j3 + m1;
  if (j3 - j1 + j2 < vmax) vmax = j3 - j1 + j2;
  if (j3 + m3 < vmax) vmax = j3 + m3;
  double C = sqrt((2.0 * j3 + 1.0) * dfact(j3 + j1 - j2) * dfact(j3 - j1 + j2) *
                  dfact(j1 + j2 - j3) * dfact(j3 + m3) * dfact(j3 - m3) /
                  (dfact(j1 + j2 + j3 + 1) * dfact(j1 - m1) * dfact(j1 + m1) *
                   dfact(j2 - m2) * dfact(j2 + m2)));
  double S = 0.0;
  for (int v = vmin; v <= vmax; ++v) {
    int par = ((v + j2 + m2) % 2 + 2) % 2;
    double t = dfact(j2 + j3 + m1 - v) * dfact(j1 - m1 + v) /
               (dfact(v) * dfact(j3 - j1 + j2 - v) * dfact(j3 + m3 - v) *
                dfact(v + j1 - j2 - m3));
    S += par ? -t : t;
  }
  return C * S;
}

static __device__ void fill_r2c(int l, cpx* Q) {
  int d = 2 * l + 1;
  for (int i = 0; i < d * d; ++i) { Q[i].re = 0.0; Q[i].im = 0.0; }
  double s = 1.0 / sqrt(2.0);
  for (int m = -l; m < 0; ++m) {
    Q[(l + m) * d + (l - m)].re = s;
    Q[(l + m) * d + (l + m)].im = -s;
  }
  Q[l * d + l].re = 1.0;
  for (int m = 1; m <= l; ++m) {
    double sg = (m & 1) ? -1.0 : 1.0;
    Q[(l + m) * d + (l + m)].re = sg * s;
    Q[(l + m) * d + (l - m)].im = sg * s;
  }
  cpx f;
  switch (l & 3) {  // (-i)^l
    case 0: f.re = 1;  f.im = 0;  break;
    case 1: f.re = 0;  f.im = -1; break;
    case 2: f.re = -1; f.im = 0;  break;
    default: f.re = 0; f.im = 1;  break;
  }
  for (int i = 0; i < d * d; ++i) Q[i] = cmul(f, Q[i]);
}

__global__ void cg_build_kernel(float* CGd) {
  __shared__ cpx Q1[49], Q2[49], Q3[169];
  __shared__ cpx Cc[7 * 7 * 13];
  int want = blockIdx.x, idx = 0, l1 = -1, l2 = -1, l3 = -1;
  for (int dout = 0; dout < 4; ++dout)
    for (int di = 0; di < 4; ++di) {
      int a = dout > di ? dout - di : di - dout;
      for (int J = a; J <= dout + di; ++J) {
        if (idx == want) { l1 = dout; l2 = di; l3 = J; }
        ++idx;
      }
    }
  if (l1 < 0) return;
  int d1 = 2 * l1 + 1, d2 = 2 * l2 + 1, d3 = 2 * l3 + 1;
  if (threadIdx.x == 0) {
    fill_r2c(l1, Q1); fill_r2c(l2, Q2); fill_r2c(l3, Q3);
    for (int i = 0; i < d1 * d2 * d3; ++i) { Cc[i].re = 0.0; Cc[i].im = 0.0; }
    for (int i = 0; i < d1; ++i)
      for (int k = 0; k < d2; ++k) {
        int m1 = i - l1, m2 = k - l2, m3 = m1 + m2;
        if (m3 >= -l3 && m3 <= l3)
          Cc[(i * d2 + k) * d3 + (l3 + m3)].re = su2cg(l1, l2, l3, m1, m2, m3);
      }
  }
  __syncthreads();
  int tot = d1 * d2 * d3;
  for (int t = threadIdx.x; t < tot; t += blockDim.x) {
    int j = t / (d2 * d3);
    int rem = t % (d2 * d3);
    int lI = rem / d3;
    int m = rem % d3;
    cpx s; s.re = 0.0; s.im = 0.0;
    for (int i = 0; i < d1; ++i)
      for (int k = 0; k < d2; ++k)
        for (int n = 0; n < d3; ++n) {
          cpx c = Cc[(i * d2 + k) * d3 + n];
          if (c.re == 0.0 && c.im == 0.0) continue;
          cpx p = cmul(cmul(Q1[i * d1 + j], Q2[k * d2 + lI]),
                       cmul(cconj(Q3[n * d3 + m]), c));
          s.re += p.re; s.im += p.im;
        }
    CGd[cg_idx(l1, l2, l3, j, lI, m)] = (float)s.re;
  }
}

// ------------------------------- KNN ---------------------------------------
__global__ void knn_kernel(const float* __restrict__ pos, int* __restrict__ nbr) {
  int n = blockIdx.x * blockDim.x + threadIdx.x;
  if (n >= N_NODES) return;
  int g0 = (n / NPG) * NPG;
  float px = pos[n * 3], py = pos[n * 3 + 1], pz = pos[n * 3 + 2];
  float bd[KNN]; int bi[KNN];
#pragma unroll
  for (int k = 0; k < KNN; ++k) { bd[k] = 3.4e38f; bi[k] = -1; }
  for (int j = g0; j < g0 + NPG; ++j) {
    if (j == n) continue;
    float dx = pos[j * 3] - px, dy = pos[j * 3 + 1] - py, dz = pos[j * 3 + 2] - pz;
    float d2 = dx * dx + dy * dy + dz * dz;
    if (d2 < bd[KNN - 1]) {
      int p = KNN - 1;
      while (p > 0 && bd[p - 1] > d2) { bd[p] = bd[p - 1]; bi[p] = bi[p - 1]; --p; }
      bd[p] = d2; bi[p] = j;
    }
  }
  for (int k = 0; k < KNN; ++k) nbr[n * KNN + k] = bi[k];
}

// ---------------------- edge geometry + spherical Y ------------------------
static __device__ float ffact(int n) {
  float f = 1.0f;
  for (int i = 2; i <= n; ++i) f *= (float)i;
  return f;
}

__global__ void edge_geom_kernel(const float* __restrict__ pos,
                                 const int* __restrict__ nbr,
                                 float* __restrict__ rel,
                                 float* __restrict__ feat,
                                 float* __restrict__ Ysh) {
  int e = blockIdx.x * blockDim.x + threadIdx.x;
  if (e >= NEDGE) return;
  int n = e / KNN;
  int j = nbr[e];
  float rx = pos[j * 3] - pos[n * 3];
  float ry = pos[j * 3 + 1] - pos[n * 3 + 1];
  float rz = pos[j * 3 + 2] - pos[n * 3 + 2];
  float r = sqrtf(rx * rx + ry * ry + rz * rz + 1e-12f);
  rel[e * 3] = rx; rel[e * 3 + 1] = ry; rel[e * 3 + 2] = rz;
  feat[e * 2] = r; feat[e * 2 + 1] = 1.0f;
  float inv = 1.0f / r;
  float dx = rx * inv, dy = ry * inv, dz = rz * inv;
  float ct = dz;
  if (ct > 1.0f) ct = 1.0f;
  if (ct < -1.0f) ct = -1.0f;
  float st2 = 1.0f - ct * ct;
  if (st2 < 1e-12f) st2 = 1e-12f;
  float st = sqrtf(st2);
  float phi = atan2f(dy, dx);
  float P[LMAX + 1][LMAX + 1];
  P[0][0] = 1.0f;
  for (int m = 1; m <= LMAX; ++m) P[m][m] = -(2.0f * m - 1.0f) * st * P[m - 1][m - 1];
  for (int m = 0; m < LMAX; ++m) P[m + 1][m] = (2.0f * m + 1.0f) * ct * P[m][m];
  for (int m = 0; m <= LMAX; ++m)
    for (int l = m + 2; l <= LMAX; ++l)
      P[l][m] = ((2.0f * l - 1.0f) * ct * P[l - 1][m] -
                 (float)(l + m - 1) * P[l - 2][m]) / (float)(l - m);
  const float FPI = 12.566370614359172f;
  for (int l = 0; l <= LMAX; ++l)
    for (int m = -l; m <= l; ++m) {
      int am = m < 0 ? -m : m;
      float Kn = sqrtf((2.0f * l + 1.0f) / FPI * ffact(l - am) / ffact(l + am));
      float v;
      if (m < 0) v = 1.4142135623730951f * Kn * sinf(am * phi) * P[l][am];
      else if (m == 0) v = Kn * P[l][0];
      else v = 1.4142135623730951f * Kn * cosf(m * phi) * P[l][m];
      Ysh[(size_t)e * 49 + l * l + l + m] = v;
    }
}

__global__ void init_h0_kernel(const float* __restrict__ x, float* __restrict__ H0) {
  int t = blockIdx.x * blockDim.x + threadIdx.x;
  if (t >= N_NODES * 3) return;
  int n = t / 3, c = t % 3;
  H0[(size_t)n * 32 + c] = x[t];  // deg-0 stride: 32 channels * (m=1)
}

// ------------------------- pair-conv message kernel ------------------------
struct PmsgArgs {
  const int* nbr;
  const float* rel;
  const float* feat;
  const float* Ysh;
  const float* CG;
  const float *w1, *b1, *w2, *b2, *w3, *b3;
  const float* Hsrc;   // source degree buffer (node stride 32*(2di+1)); null if rel-only
  float* acc;          // [edge][c_out][2*dout+1]
  int di, dout, c_in, c_out, nJ;
  int src_mode;        // 0: Hsrc; 1: Hsrc + rel on ch0; 2: rel only (c_in==1)
  int init;            // 1: overwrite, 0: accumulate
};

__global__ void __launch_bounds__(128) pmsg_kernel(PmsgArgs A) {
  extern __shared__ float sm[];
  const int di = A.di, dout = A.dout, cin = A.c_in, nJ = A.nJ;
  const int idim = 2 * di + 1, odim = 2 * dout + 1;
  const int outc = cin * nJ;            // w3 columns per output channel a
  const int Jmin = (di > dout) ? di - dout : dout - di;

  float* sm_feat = sm;                          // 16*32
  float* sm_w1   = sm_feat + 16 * 32;           // 32*32 (rows>=2 zero)
  float* sm_w2   = sm_w1 + 32 * 32;             // 32*32
  float* sm_h1   = sm_w2 + 32 * 32;             // 16*32
  float* sm_h2   = sm_h1 + 16 * 32;             // 16*32
  float* sm_f    = sm_h2 + 16 * 32;             // 16*cin*idim
  float* sm_B    = sm_f + 16 * cin * idim;      // 16*odim*idim*nJ
  float* sm_T    = sm_B + 16 * odim * idim * nJ;// 16*odim*cin*nJ
  float* sm_R    = sm_T + 16 * odim * cin * nJ; // 16*outc
  float* sm_w3   = sm_R + 16 * outc;            // 32*outc (async-staged panel, last)

  const int tid = threadIdx.x;
  const int wave = tid >> 5;
  const int e0 = blockIdx.x * 16;

  // --- stage feat (K-padded), w1 (K-padded), w2 ---
  for (int i = tid; i < 16 * 32; i += 128) {
    int e = i >> 5, c = i & 31;
    sm_feat[i] = (c < 2) ? A.feat[(size_t)(e0 + e) * 2 + c] : 0.0f;
  }
  for (int i = tid; i < 32 * 32; i += 128) {
    int r = i >> 5;
    sm_w1[i] = (r < 2) ? A.w1[i] : 0.0f;
    sm_w2[i] = A.w2[i];
  }
  // --- gather source features ---
  {
    int ftot = 16 * cin * idim;
    for (int i = tid; i < ftot; i += 128) {
      int e = i / (cin * idim);
      int rem = i - e * (cin * idim);
      int c = rem / idim;
      int m = rem - c * idim;
      int eg = e0 + e;
      float v;
      if (A.src_mode == 2) {
        v = A.rel[(size_t)eg * 3 + m];
      } else {
        int node = A.nbr[eg];
        v = A.Hsrc[(size_t)node * (32 * idim) + c * idim + m];
        if (A.src_mode == 1 && c == 0) v += A.rel[(size_t)eg * 3 + m];
      }
      sm_f[i] = v;
    }
  }
  // --- build basis B[e][o][i][j] from CG x Y ---
  {
    int btot = 16 * odim * idim * nJ;
    for (int i = tid; i < btot; i += 128) {
      int e = i / (odim * idim * nJ);
      int rem = i - e * (odim * idim * nJ);
      int o = rem / (idim * nJ);
      int rem2 = rem - o * (idim * nJ);
      int ii = rem2 / nJ;
      int j = rem2 - ii * nJ;
      int J = Jmin + j;
      const float* Yp = A.Ysh + (size_t)(e0 + e) * 49 + J * J;
      const float* Cp = A.CG + cg_idx(dout, di, J, o, ii, 0);
      float s = 0.0f;
      for (int m = 0; m < 2 * J + 1; ++m) s += Cp[m] * Yp[m];
      sm_B[i] = s;
    }
  }
  __syncthreads();
  // --- factored tensor T[e][o][c][j] = sum_i B[e][o][i][j] * f[e][c][i] ---
  {
    int ttot = 16 * odim * cin * nJ;
    for (int i = tid; i < ttot; i += 128) {
      int e = i / (odim * cin * nJ);
      int rem = i - e * (odim * cin * nJ);
      int o = rem / (cin * nJ);
      int rem2 = rem - o * (cin * nJ);
      int c = rem2 / nJ;
      int j = rem2 - c * nJ;
      float s = 0.0f;
      for (int ii = 0; ii < idim; ++ii)
        s += sm_B[((e * odim + o) * idim + ii) * nJ + j] *
             sm_f[(e * cin + c) * idim + ii];
      sm_T[i] = s;
    }
  }
  __syncthreads();
  // --- radial MLP stage 1: feat(16x32,pad) @ w1(32x32,pad) -> relu h1 ---
  if (wave < 2) {
    v16h a = frag_a_f16(sm_feat, 32);
    v16h b = frag_b_lds(sm_w1, 32, wave * 16);
    v8f c = {};
    c = wmma_f16(a, b, c);
    store_tile(sm_h1, 32, c, wave * 16, A.b1, 0, 1, 32);
  }
  __syncthreads();
  // --- stage 2: h1 @ w2 -> relu h2 ---
  if (wave < 2) {
    v16h a = frag_a_f16(sm_h1, 32);
    v16h b = frag_b_lds(sm_w2, 32, wave * 16);
    v8f c = {};
    c = wmma_f16(a, b, c);
    store_tile(sm_h2, 32, c, wave * 16, A.b2, 0, 1, 32);
  }
  __syncthreads();

  // A fragment of the w3 GEMM is invariant across the a/tile loops: hoist.
  v16h af = frag_a_f16(sm_h2, 32);

  // prefetch w3 into GL2 (L2-resident weight panel)
  {
    size_t w3n = (size_t)32 * A.c_out * outc;
    for (size_t p = (size_t)tid * 64; p < w3n; p += (size_t)128 * 64)
      __builtin_prefetch(A.w3 + p);
  }

  // --- stage 3 per output channel a: R = h2 @ w3[:, a-slice] (+b3),
  //     then out[e][a][o] = sum_{c,j} R[e][c*nJ+j] * T[e][o][c][j] ---
  const int ntile = (outc + 15) >> 4;
  const int ldw3 = A.c_out * outc;
  const int esub = tid >> 3;        // edge within tile
  const int osub = tid & 7;         // output order index
  const unsigned w3_lds = (unsigned)(size_t)(void*)sm_w3;
  for (int a = 0; a < A.c_out; ++a) {
    // async-stage the 32 x outc weight panel into LDS (ASYNCcnt path)
    {
      int tot = 32 * outc;
      int gbase = a * outc;
      for (int q = tid; q < tot; q += 128) {
        int row = q / outc, col = q - row * outc;
        async_g2l_b32(w3_lds + (unsigned)q * 4u, A.w3,
                      (row * ldw3 + gbase + col) * 4);
      }
      wait_asynccnt0();
    }
    __syncthreads();
    for (int t = wave; t < ntile; t += 4) {
      // reads past outc on ragged tiles only feed discarded D columns
      v16h bf = frag_b_lds(sm_w3, outc, t * 16);
      v8f c = {};
      c = wmma_f16(af, bf, c);
      store_tile(sm_R, outc, c, t * 16, A.b3, a * outc, 0, outc);
    }
    __syncthreads();
    if (osub < odim) {
      const float* Rp = sm_R + esub * outc;
      const float* Tp = sm_T + ((esub * odim + osub) * cin) * nJ;
      float s = 0.0f;
      for (int cj = 0; cj < outc; ++cj) s += Rp[cj] * Tp[cj];
      size_t eg = (size_t)(e0 + esub);
      float* ap = A.acc + (eg * A.c_out + a) * odim + osub;
      if (A.init) *ap = s; else *ap += s;
    }
    // next-iteration asyncs write sm_w3 only; sm_R reuse is ordered by the
    // pre-WMMA barrier of the next iteration.
  }
}

// ------------------------------ attention ----------------------------------
struct AttnArgs {
  const float* Hin[4];
  float* Hout[4];
  const float* vacc[4];
  const float* kacc[4];
  const float* qw[4];
  const float* proj[4];
  const float* ng[4];
  const float* nb[4];
  int cprev[4];
  float inv_sqrt_kdim;
};

__global__ void __launch_bounds__(128) attn_kernel(AttnArgs A) {
  __shared__ float qv[4][HEAD_C][7];
  __shared__ float logit[KNN];
  __shared__ float alpha[KNN];
  __shared__ float agg[HEAD_C][7];
  int n = blockIdx.x, tid = threadIdx.x;
  // q[d] = qw[d] @ h[d]
  for (int d = 0; d < 4; ++d) {
    int cp = A.cprev[d];
    if (!cp) continue;
    int md = 2 * d + 1;
    for (int i = tid; i < HEAD_C * md; i += 128) {
      int o = i / md, m = i - o * md;
      const float* h = A.Hin[d] + (size_t)n * 32 * md;
      const float* w = A.qw[d] + o * cp;
      float s = 0.0f;
      for (int c = 0; c < cp; ++c) s += w[c] * h[c * md + m];
      qv[d][o][m] = s;
    }
  }
  __syncthreads();
  if (tid < KNN) {
    float s = 0.0f;
    for (int d = 0; d < 4; ++d) {
      if (!A.cprev[d]) continue;
      int md = 2 * d + 1;
      const float* kk = A.kacc[d] + ((size_t)(n * KNN + tid) * HEAD_C) * md;
      for (int o = 0; o < HEAD_C; ++o)
        for (int m = 0; m < md; ++m) s += kk[o * md + m] * qv[d][o][m];
    }
    logit[tid] = s * A.inv_sqrt_kdim;
  }
  __syncthreads();
  if (tid == 0) {
    float mx = -3.4e38f;
    for (int k = 0; k < KNN; ++k) mx = logit[k] > mx ? logit[k] : mx;
    float sum = 0.0f;
    for (int k = 0; k < KNN; ++k) { float a = expf(logit[k] - mx); alpha[k] = a; sum += a; }
    float inv = 1.0f / sum;
    for (int k = 0; k < KNN; ++k) alpha[k] *= inv;
  }
  __syncthreads();
  for (int d = 0; d < 4; ++d) {
    int md = 2 * d + 1;
    for (int i = tid; i < HEAD_C * md; i += 128) {
      int o = i / md, m = i - o * md;
      float s = 0.0f;
      for (int k = 0; k < KNN; ++k)
        s += alpha[k] * A.vacc[d][((size_t)(n * KNN + k) * HEAD_C + o) * md + m];
      agg[o][m] = s;
    }
    __syncthreads();
    if (tid < 32) {
      int c = tid;
      float y[7];
      for (int m = 0; m < md; ++m) {
        float s = 0.0f;
        for (int o = 0; o < HEAD_C; ++o) s += A.proj[d][c * HEAD_C + o] * agg[o][m];
        y[m] = s;
      }
      if (c < A.cprev[d]) {
        const float* h = A.Hin[d] + (size_t)n * 32 * md;
        for (int m = 0; m < md; ++m) y[m] += h[c * md + m];
      }
      float n2 = 1e-12f;
      for (int m = 0; m < md; ++m) n2 += y[m] * y[m];
      float nr = sqrtf(n2);
      float sc = nr * A.ng[d][c] + A.nb[d][c];
      sc = sc > 0.0f ? sc : 0.0f;
      float f = sc / nr;
      float* ho = A.Hout[d] + (size_t)n * 32 * md;
      for (int m = 0; m < md; ++m) ho[c * md + m] = y[m] * f;
    }
    __syncthreads();
  }
}

// -------------------------- final reduce + head ----------------------------
__global__ void final_reduce_kernel(const float* __restrict__ msg,
                                    const float* __restrict__ H0,
                                    const float* __restrict__ si,
                                    float* __restrict__ out0) {
  int t = blockIdx.x * blockDim.x + threadIdx.x;
  if (t >= N_NODES * 32) return;
  int n = t / 32, c = t - n * 32;
  float s = 0.0f;
  for (int k = 0; k < KNN; ++k) s += msg[((size_t)(n * KNN + k) * 32) + c];
  s *= (1.0f / KNN);
  const float* h = H0 + (size_t)n * 32;
  float s2 = 0.0f;
  for (int c2 = 0; c2 < 32; ++c2) s2 += si[c * 32 + c2] * h[c2];
  out0[t] = s + s2;
}

__global__ void __launch_bounds__(512) pool_fc_kernel(const float* __restrict__ out0,
                                                      const float* __restrict__ w1,
                                                      const float* __restrict__ b1,
                                                      const float* __restrict__ w2,
                                                      const float* __restrict__ b2,
                                                      float* __restrict__ out) {
  __shared__ float g[GRAPHS][32];
  __shared__ float z[GRAPHS][32];
  int tid = threadIdx.x;
  if (tid < GRAPHS * 32) {
    int gr = tid >> 5, c = tid & 31;
    float m = -3.4e38f;
    for (int i = 0; i < NPG; ++i) {
      float v = out0[(size_t)(gr * NPG + i) * 32 + c];
      m = v > m ? v : m;
    }
    g[gr][c] = m;
  }
  __syncthreads();
  if (tid < GRAPHS * 32) {
    int gr = tid >> 5, o = tid & 31;
    float s = b1[o];
    for (int i = 0; i < 32; ++i) s += g[gr][i] * w1[i * 32 + o];
    z[gr][o] = s > 0.0f ? s : 0.0f;
  }
  __syncthreads();
  for (int idx = tid; idx < GRAPHS * 512; idx += 512) {
    int gr = idx >> 9, o = idx & 511;
    float s = b2[o];
    for (int i = 0; i < 32; ++i) s += z[gr][i] * w2[i * 512 + o];
    out[idx] = s;
  }
}

// ------------------------------- host side ---------------------------------
struct PC { const float *b1, *b2, *b3, *w1, *w2, *w3; };

static size_t pmsg_lds_bytes(int di, int dout, int cin, int nJ) {
  int idim = 2 * di + 1, odim = 2 * dout + 1, outc = cin * nJ;
  size_t fl = (size_t)16 * 32 + 32 * 32 + 32 * 32 + 16 * 32 + 16 * 32 +
              (size_t)16 * cin * idim + (size_t)16 * odim * idim * nJ +
              (size_t)16 * odim * cin * nJ + (size_t)16 * outc +
              (size_t)32 * outc;
  return fl * sizeof(float);
}

extern "C" void kernel_launch(void* const* d_in, const int* in_sizes, int n_in,
                              void* d_out, int out_size, void* d_ws, size_t ws_size,
                              hipStream_t stream) {
  (void)in_sizes; (void)n_in; (void)out_size; (void)ws_size;
  const float* x   = (const float*)d_in[0];
  const float* pos = (const float*)d_in[1];
  // d_in[2] = batch (contiguous graphs of 512 nodes; derived analytically)

  // -------- parameter leaf extraction (JAX pytree order: dicts key-sorted) --
  int ci = 3;
  auto nxt = [&]() { return (const float*)d_in[ci++]; };
  auto readPC = [&]() {
    PC p; p.b1 = nxt(); p.b2 = nxt(); p.b3 = nxt();
    p.w1 = nxt(); p.w2 = nxt(); p.w3 = nxt(); return p;
  };
  // 'fc'
  const float* fc_b1 = nxt(); const float* fc_b2 = nxt();
  const float* fc_w1 = nxt(); const float* fc_w2 = nxt();
  // 'final': 'conv'{0..3}, 'si'
  PC fconv[4];
  for (int d = 0; d < 4; ++d) fconv[d] = readPC();
  const float* fsi = nxt();
  // 'layers'
  struct LayerP {
    PC k[4][4]; const float* nb[4]; const float* ng[4];
    const float* proj[4]; const float* q[4]; PC v[4][4];
  } LP[4];
  for (int L = 0; L < 4; ++L) {
    int nsrc = (L == 0) ? 2 : 4;
    int kqn  = (L == 0) ? 1 : 4;
    for (int di = 0; di < nsrc; ++di)
      for (int dout = 0; dout < kqn; ++dout) LP[L].k[di][dout] = readPC();
    for (int d = 0; d < 4; ++d) LP[L].nb[d] = nxt();
    for (int d = 0; d < 4; ++d) LP[L].ng[d] = nxt();
    for (int d = 0; d < 4; ++d) LP[L].proj[d] = nxt();
    for (int d = 0; d < 4; ++d) LP[L].q[d] = nullptr;
    for (int d = 0; d < kqn; ++d) LP[L].q[d] = nxt();
    for (int di = 0; di < nsrc; ++di)
      for (int dout = 0; dout < 4; ++dout) LP[L].v[di][dout] = readPC();
  }

  // -------- workspace carve (fixed deterministic layout, ~52 MB) -----------
  char* ws = (char*)d_ws;
  size_t off = 0;
  auto alloc = [&](size_t bytes) {
    void* p = ws + off;
    off += (bytes + 255) & ~(size_t)255;
    return p;
  };
  int*   nbr  = (int*)alloc((size_t)N_NODES * KNN * 4);
  float* rel  = (float*)alloc((size_t)NEDGE * 3 * 4);
  float* feat = (float*)alloc((size_t)NEDGE * 2 * 4);
  float* Ysh  = (float*)alloc((size_t)NEDGE * 49 * 4);
  float* CGd  = (float*)alloc((size_t)4 * 4 * 7 * 7 * 7 * 13 * 4);
  size_t hbytes = (size_t)N_NODES * 32 * 16 * 4;  // sum over degrees of 32*(2d+1)
  float* hA = (float*)alloc(hbytes);
  float* hB = (float*)alloc(hbytes);
  float* vacc[4];
  float* kacc[4];
  for (int d = 0; d < 4; ++d) vacc[d] = (float*)alloc((size_t)NEDGE * HEAD_C * (2 * d + 1) * 4);
  for (int d = 0; d < 4; ++d) kacc[d] = (float*)alloc((size_t)NEDGE * HEAD_C * (2 * d + 1) * 4);
  float* msg  = (float*)alloc((size_t)NEDGE * 32 * 4);
  float* out0 = (float*)alloc((size_t)N_NODES * 32 * 4);

  size_t doffF[4] = {0, (size_t)N_NODES * 32 * 1, (size_t)N_NODES * 32 * 4,
                     (size_t)N_NODES * 32 * 9};  // cumulative 32*(2d+1) per node

  hipFuncSetAttribute((const void*)pmsg_kernel,
                      hipFuncAttributeMaxDynamicSharedMemorySize, 224 * 1024);

  // -------- graph construction + bases -------------------------------------
  hipMemsetAsync(hA, 0, hbytes, stream);
  cg_build_kernel<<<44, 128, 0, stream>>>(CGd);
  knn_kernel<<<N_NODES / 128, 128, 0, stream>>>(pos, nbr);
  edge_geom_kernel<<<NEDGE / 256, 256, 0, stream>>>(pos, nbr, rel, feat, Ysh);
  init_h0_kernel<<<(N_NODES * 3 + 255) / 256, 256, 0, stream>>>(x, hA);

  auto launch_pmsg = [&](const PC& pc, int di, int dout, int cin, int cout, int nJ,
                         const float* Hsrc, int mode, float* acc, int init) {
    PmsgArgs A;
    A.nbr = nbr; A.rel = rel; A.feat = feat; A.Ysh = Ysh; A.CG = CGd;
    A.w1 = pc.w1; A.b1 = pc.b1; A.w2 = pc.w2; A.b2 = pc.b2;
    A.w3 = pc.w3; A.b3 = pc.b3;
    A.Hsrc = Hsrc; A.acc = acc;
    A.di = di; A.dout = dout; A.c_in = cin; A.c_out = cout; A.nJ = nJ;
    A.src_mode = mode; A.init = init;
    size_t lds = pmsg_lds_bytes(di, dout, cin, nJ);
    pmsg_kernel<<<NEDGE / 16, 128, lds, stream>>>(A);
  };

  // -------- layers ----------------------------------------------------------
  float* hbuf[2] = {hA, hB};
  int cprev[4] = {3, 0, 0, 0};
  for (int L = 0; L < 4; ++L) {
    float* Hi = hbuf[L & 1];
    float* Ho = hbuf[(L + 1) & 1];
    int nsrc = (L == 0) ? 2 : 4;
    int kqn  = (L == 0) ? 1 : 4;
    int kdim = 0;
    for (int d = 0; d < kqn; ++d) kdim += HEAD_C * (2 * d + 1);
    // value messages (all output degrees)
    for (int dout = 0; dout < 4; ++dout)
      for (int di = 0; di < nsrc; ++di) {
        int cin, mode; const float* Hsrc;
        if (L == 0) {
          if (di == 0) { cin = 3; mode = 0; Hsrc = Hi + doffF[0]; }
          else         { cin = 1; mode = 2; Hsrc = nullptr; }
        } else { cin = 32; mode = (di == 1) ? 1 : 0; Hsrc = Hi + doffF[di]; }
        int mn = di < dout ? di : dout;
        launch_pmsg(LP[L].v[di][dout], di, dout, cin, HEAD_C, 2 * mn + 1,
                    Hsrc, mode, vacc[dout], di == 0);
      }
    // key messages (output degrees present in input h)
    for (int dout = 0; dout < kqn; ++dout)
      for (int di = 0; di < nsrc; ++di) {
        int cin, mode; const float* Hsrc;
        if (L == 0) {
          if (di == 0) { cin = 3; mode = 0; Hsrc = Hi + doffF[0]; }
          else         { cin = 1; mode = 2; Hsrc = nullptr; }
        } else { cin = 32; mode = (di == 1) ? 1 : 0; Hsrc = Hi + doffF[di]; }
        int mn = di < dout ? di : dout;
        launch_pmsg(LP[L].k[di][dout], di, dout, cin, HEAD_C, 2 * mn + 1,
                    Hsrc, mode, kacc[dout], di == 0);
      }
    // attention + proj + residual + norm
    AttnArgs AA;
    for (int d = 0; d < 4; ++d) {
      AA.Hin[d]  = Hi + doffF[d];
      AA.Hout[d] = Ho + doffF[d];
      AA.vacc[d] = vacc[d];
      AA.kacc[d] = kacc[d];
      AA.qw[d]   = LP[L].q[d];
      AA.proj[d] = LP[L].proj[d];
      AA.ng[d]   = LP[L].ng[d];
      AA.nb[d]   = LP[L].nb[d];
      AA.cprev[d] = cprev[d];
    }
    AA.inv_sqrt_kdim = 1.0f / sqrtf((float)kdim);
    attn_kernel<<<N_NODES, 128, 0, stream>>>(AA);
    for (int d = 0; d < 4; ++d) cprev[d] = 32;
  }

  // -------- final invariant conv + pooled head ------------------------------
  float* Hf = hbuf[0];  // after 4 layers, h lives in buffer A
  for (int di = 0; di < 4; ++di)
    launch_pmsg(fconv[di], di, 0, 32, 32, 1, Hf + doffF[di], 0, msg, di == 0);
  final_reduce_kernel<<<(N_NODES * 32) / 256, 256, 0, stream>>>(msg, Hf + doffF[0],
                                                                fsi, out0);
  pool_fc_kernel<<<1, 512, 0, stream>>>(out0, fc_w1, fc_b1, fc_w2, fc_b2,
                                        (float*)d_out);
}
